// LightResidualHead_790273982956
// MI455X (gfx1250) — matrix operand order repaired and checked by reading, not compiled, not run on hardware
//
#include <hip/hip_runtime.h>

typedef __attribute__((ext_vector_type(16))) _Float16 v16h;
typedef __attribute__((ext_vector_type(8)))  float    v8f;

#define D 128
#define ROWS_PER_WAVE 16
#define WAVES_PER_WG 8
#define ROWS_PER_WG (ROWS_PER_WAVE * WAVES_PER_WG)

// ---------------------------------------------------------------------------
// Algebraic folding (exact):
//   z_ref = x_hat*g + b
//   h1    = GELU(z_ref @ w1 + b1) = GELU(x_hat @ W1' + B1')
//     W1' = diag(g) w1,  B1' = b1 + b^T w1
//   out   = (z_ref + h1@w2 + b2) @ w_out + b_out
//         = x_hat @ (g*w_out) + h1 @ (w2 @ w_out) + (b^T w_out + b2^T w_out + b_out)
// So only ONE 128x128 GEMM (W1') is needed; everything else is vectors.
//
// Prep lays W1' out as V_WMMA_F32_16X16X32_F16 B fragments in this kernel's own k order (A matches):
//   frag f = kt*8+nt; lane l: N = nt*16+(l&15), Kbase = kt*32+(l&16?16:0);
//   element e -> k = Kbase+e (VGPR v holds K=2v,2v+1 packed).
// g*w_out is stored as N-REPLICATED fragments so a WMMA produces the row-dot
// x_hat@ (g*w_out) directly in every lane's D fragment (no reduction needed).
// ---------------------------------------------------------------------------
__global__ __launch_bounds__(256)
void prep_weights(const float* __restrict__ w1, const float* __restrict__ w2,
                  const float* __restrict__ gamma, const float* __restrict__ beta,
                  const float* __restrict__ b1, const float* __restrict__ b2,
                  const float* __restrict__ w_out, const float* __restrict__ b_out,
                  _Float16* __restrict__ f1, _Float16* __restrict__ fwog,
                  float* __restrict__ bias1, float* __restrict__ wv2,
                  float* __restrict__ outb) {
  int tid = blockIdx.x * blockDim.x + threadIdx.x;   // 0..16383
  if (tid < D * D) {
    int e    = tid & 15;
    int lane = (tid >> 4) & 31;
    int f    = tid >> 9;
    int kt   = f >> 3;
    int nt   = f & 7;
    int n = nt * 16 + (lane & 15);
    int k = kt * 32 + ((lane & 16) ? 16 : 0) + e;
    f1[tid] = (_Float16)(gamma[k] * w1[k * D + n]);      // gamma folded into w1
  }
  if (tid < 4 * 512) {               // g*w_out fragments, replicated over N
    int e    = tid & 15;
    int lane = (tid >> 4) & 31;
    int kt   = tid >> 9;
    int k = kt * 32 + ((lane & 16) ? 16 : 0) + e;
    fwog[tid] = (_Float16)(gamma[k] * w_out[k]);
  }
  if (blockIdx.x == 0) {
    int t = threadIdx.x;
    if (t < D) {
      float s = b1[t];                                   // B1' = b1 + beta^T w1
      for (int k = 0; k < D; ++k) s += beta[k] * w1[k * D + t];
      bias1[t] = s;
      wv2[t] = w2[t * D] * w_out[0];                     // wv2 = w2 @ w_out
      for (int e = 1; e < D; ++e) wv2[t] += w2[t * D + e] * w_out[e];
    } else if (t == D) {           // scalar: beta^T w_out + b2^T w_out + b_out
      float c = b_out[0];
      for (int k = 0; k < D; ++k) c += (beta[k] + b2[k]) * w_out[k];
      outb[0] = c;
    }
  }
}

// tanh-form GELU: one CDNA5 v_tanh_f32 transcendental, branchless.
__device__ __forceinline__ float gelu_fast(float v) {
  float u = v * __builtin_fmaf(0.0356774081f, v * v, 0.7978845608f);
  float t;
#if __has_builtin(__builtin_amdgcn_tanhf)
  t = __builtin_amdgcn_tanhf(u);
#else
  asm volatile("v_tanh_f32 %0, %1\n\tv_nop" : "=v"(t) : "v"(u));
#endif
  float hv = 0.5f * v;
  return __builtin_fmaf(hv, t, hv);
}

// ---------------------------------------------------------------------------
// Fused: LayerNorm -> WMMA GEMM1 + GELU -> folded projections -> out
// One wave privately owns 16 rows; 8 waves/WG; no barriers after weight stage.
// ---------------------------------------------------------------------------
__global__ __launch_bounds__(256)
void fused_ln_mlp_head(const float* __restrict__ x,
                       const _Float16* __restrict__ w1f,
                       const float* __restrict__ bias1,
                       const _Float16* __restrict__ fwog,
                       const float* __restrict__ wv2,
                       const float* __restrict__ outb,
                       float* __restrict__ out) {
  // zbuf: f32 x_hat, XOR-swizzled on 16B chunks by row (bank-conflict-free).
  __shared__ __align__(16) float     zbuf[WAVES_PER_WG * ROWS_PER_WAVE * D]; // 64 KB
  __shared__ __align__(16) _Float16  w1l[D * D];                             // 32 KB

  {// stage pre-swizzled W1' fragments into LDS (source is L2-hot)
    int t = threadIdx.x;
    float4*       d1 = (float4*)w1l;
    const float4* s1 = (const float4*)w1f;
    #pragma unroll
    for (int i = 0; i < 8; ++i) d1[t + i * 256] = s1[t + i * 256];
  }
  __syncthreads();

  const int lane = threadIdx.x & 31;
  const int wv   = threadIdx.x >> 5;
  const int m    = lane & 15;
  const int hi   = (lane >> 4) & 1;
  const int koff = hi * 16;           // A-fragment K-run offset
  const int mb   = hi * 8;            // C/D fragment M base

  const long rowBase = (long)blockIdx.x * ROWS_PER_WG + (long)wv * ROWS_PER_WAVE;

  float* zt = &zbuf[wv * ROWS_PER_WAVE * D];

  // ---------------- LayerNorm: lane = (row m, 64-col half hi) ---------------
  {
    const float* xrow = x + (rowBase + m) * D + hi * 64;
    float4 xv[16];
    #pragma unroll
    for (int i = 0; i < 16; ++i) xv[i] = *(const float4*)(xrow + 4 * i);

    float s = 0.f, ss = 0.f;
    #pragma unroll
    for (int i = 0; i < 16; ++i) {
      float4 v = xv[i];
      s  += (v.x + v.y) + (v.z + v.w);
      ss  = __builtin_fmaf(v.x, v.x, ss);
      ss  = __builtin_fmaf(v.y, v.y, ss);
      ss  = __builtin_fmaf(v.z, v.z, ss);
      ss  = __builtin_fmaf(v.w, v.w, ss);
    }
    s  += __shfl_xor(s, 16, 32);      // combine the two 64-col halves
    ss += __shfl_xor(ss, 16, 32);
    float mu   = s * (1.0f / D);
    float var  = ss * (1.0f / D) - mu * mu;
    float rstd = rsqrtf(var + 1e-5f);
    #pragma unroll
    for (int i = 0; i < 16; ++i) {
      float4 v = xv[i];
      float4 z;
      z.x = (v.x - mu) * rstd; z.y = (v.y - mu) * rstd;
      z.z = (v.z - mu) * rstd; z.w = (v.w - mu) * rstd;
      int c = hi * 16 + i;                         // 16B chunk index 0..31
      *(float4*)(zt + m * D + ((c ^ m) * 4)) = z;  // XOR swizzle by row
    }
  }

  // ---------------- A fragments (f32 LDS -> f16, same swizzle) --------------
  v16h a1[4];
  #pragma unroll
  for (int kt = 0; kt < 4; ++kt) {
    int c0 = kt * 8 + (koff >> 2);
    const float* zr = zt + m * D;
    float4 q0 = *(const float4*)(zr + ((c0      ^ m) * 4));
    float4 q1 = *(const float4*)(zr + (((c0 + 1) ^ m) * 4));
    float4 q2 = *(const float4*)(zr + (((c0 + 2) ^ m) * 4));
    float4 q3 = *(const float4*)(zr + (((c0 + 3) ^ m) * 4));
    v16h a;
    a[0]=(_Float16)q0.x; a[1]=(_Float16)q0.y; a[2]=(_Float16)q0.z; a[3]=(_Float16)q0.w;
    a[4]=(_Float16)q1.x; a[5]=(_Float16)q1.y; a[6]=(_Float16)q1.z; a[7]=(_Float16)q1.w;
    a[8]=(_Float16)q2.x; a[9]=(_Float16)q2.y; a[10]=(_Float16)q2.z; a[11]=(_Float16)q2.w;
    a[12]=(_Float16)q3.x; a[13]=(_Float16)q3.y; a[14]=(_Float16)q3.z; a[15]=(_Float16)q3.w;
    a1[kt] = a;
  }

  const v16h* w1v  = (const v16h*)w1l;
  const v16h* wogv = (const v16h*)fwog;   // replicated-N fragments (L2-hot)

  // ---- residual projection: x_hat @ (g*w_out) via WMMA (all N cols equal) --
  v8f zproj = {};
  #pragma unroll
  for (int kt = 0; kt < 4; ++kt)
    zproj = __builtin_amdgcn_wmma_f32_16x16x32_f16(false, a1[kt], false,
                                                   wogv[kt * 32 + lane],
                                                   (short)0, zproj, false, false);

  // ---------------- GEMM1 + bias + GELU, fold wv2 projection ----------------
  float pacc[8];
  #pragma unroll
  for (int j = 0; j < 8; ++j) pacc[j] = 0.0f;

  #pragma unroll
  for (int nt = 0; nt < 8; ++nt) {
    int   q    = nt * 16 + m;          // GEMM1 output feature index
    float bias = bias1[q];
    v8f acc;
    #pragma unroll
    for (int j = 0; j < 8; ++j) acc[j] = bias;
    #pragma unroll
    for (int kt = 0; kt < 4; ++kt)
      acc = __builtin_amdgcn_wmma_f32_16x16x32_f16(false, a1[kt], false,
                                                   w1v[(kt * 8 + nt) * 32 + lane],
                                                   (short)0, acc, false, false);
    float wq = wv2[q];                 // (w2 @ w_out)[q]
    #pragma unroll
    for (int j = 0; j < 8; ++j)
      pacc[j] = __builtin_fmaf(gelu_fast(acc[j]), wq, pacc[j]);
  }

  // cross-lane sum over the 16 feature columns of each half
  #pragma unroll
  for (int dxy = 1; dxy < 16; dxy <<= 1) {
    #pragma unroll
    for (int j = 0; j < 8; ++j) pacc[j] += __shfl_xor(pacc[j], dxy, 32);
  }

  if (m == 0) {                        // lanes 0 and 16 write 8 rows each
    float bo = outb[0];
    #pragma unroll
    for (int j = 0; j < 8; ++j)
      out[rowBase + mb + j] = pacc[j] + zproj[j] + bo;
  }
}

// ---------------------------------------------------------------------------
extern "C" void kernel_launch(void* const* d_in, const int* in_sizes, int n_in,
                              void* d_out, int out_size, void* d_ws, size_t ws_size,
                              hipStream_t stream) {
  const float* x     = (const float*)d_in[0];
  const float* gamma = (const float*)d_in[1];
  const float* beta  = (const float*)d_in[2];
  const float* w1    = (const float*)d_in[3];
  const float* b1    = (const float*)d_in[4];
  const float* w2    = (const float*)d_in[5];
  const float* b2    = (const float*)d_in[6];
  const float* w_o   = (const float*)d_in[7];
  const float* b_o   = (const float*)d_in[8];
  float* out = (float*)d_out;

  _Float16* f1    = (_Float16*)d_ws;            // 16384 halves (W1' frags)
  _Float16* fwog  = f1 + D * D;                 // 2048 halves  (g*w_out frags)
  float*    fbuf  = (float*)(fwog + 2048);      // f32 region
  float*    bias1 = fbuf;                       // 128  (B1')
  float*    wv2   = fbuf + 128;                 // 128  (w2 @ w_out)
  float*    outb  = fbuf + 256;                 // 1    (folded output bias)

  prep_weights<<<(D * D + 255) / 256, 256, 0, stream>>>(
      w1, w2, gamma, beta, b1, b2, w_o, b_o, f1, fwog, bias1, wv2, outb);

  long rows  = (long)in_sizes[0] / D;
  int  tiles = (int)(rows / ROWS_PER_WG);
  fused_ln_mlp_head<<<tiles, 256, 0, stream>>>(x, f1, bias1, fwog, wv2, outb,
                                               out);
}
